// UpgradedBINN_50646254354906
// MI455X (gfx1250) — compile-verified, matrix operand order checked by reading, not run on hardware
//
#include <hip/hip_runtime.h>
#include <hip/hip_bf16.h>

typedef unsigned short u16t;
typedef __attribute__((ext_vector_type(16))) __bf16 v16bf;
typedef __attribute__((ext_vector_type(8)))  float  v8f;

#define BN_EPS 1e-3f

__device__ __forceinline__ u16t f2bf(float f) {
    unsigned u = __float_as_uint(f);
    return (u16t)((u + 0x7FFFu + ((u >> 16) & 1u)) >> 16);   // round-to-nearest-even
}

// CDNA5 async global->LDS copy (ASYNCcnt-tracked, no VGPR staging).
// vdst = per-lane LDS byte address, vaddr = per-lane 64-bit global address.
__device__ __forceinline__ void async_copy_b128(const void* gptr, unsigned lds_byte_addr) {
    asm volatile("global_load_async_to_lds_b128 %0, %1, off"
                 :
                 : "v"(lds_byte_addr), "v"(gptr)
                 : "memory");
}
__device__ __forceinline__ void async_wait0() {
    asm volatile("s_wait_asynccnt 0" ::: "memory");
}

// ---------------- prep kernels ----------------

// out_bf16[i] = bf16(a[i]); n4 = count/4  (x conversion)
__global__ void k_cvt(const float* __restrict__ a, u16t* __restrict__ out, long long n4) {
    long long i = (long long)blockIdx.x * blockDim.x + threadIdx.x;
    if (i >= n4) return;
    float4 av = ((const float4*)a)[i];
    uint2 r;
    r.x = (unsigned)f2bf(av.x) | ((unsigned)f2bf(av.y) << 16);
    r.y = (unsigned)f2bf(av.z) | ((unsigned)f2bf(av.w) << 16);
    ((uint2*)out)[i] = r;
}

// outT[n, k] = bf16(W[k, n] * Mk[k, n]) for n < N, else 0.
// W, Mk: [K x N] f32 row-major.  outT: [Np x K] bf16 row-major (Np = padded N).
__global__ __launch_bounds__(256)
void k_mulcvt_T(const float* __restrict__ W, const float* __restrict__ Mk,
                u16t* __restrict__ outT, int K, int N) {
    __shared__ u16t tile[32][33];
    const int tx = threadIdx.x;          // 0..31
    const int ty = threadIdx.y;          // 0..7
    const int n0 = blockIdx.x * 32;
    const int k0 = blockIdx.y * 32;
#pragma unroll
    for (int i = 0; i < 4; ++i) {
        int k = k0 + ty + i * 8;
        int n = n0 + tx;
        u16t v = 0;
        if (n < N) {
            size_t idx = (size_t)k * N + n;
            v = f2bf(W[idx] * Mk[idx]);
        }
        tile[tx][ty + i * 8] = v;        // tile[n_local][k_local]
    }
    __syncthreads();
#pragma unroll
    for (int i = 0; i < 4; ++i) {
        int nl = ty + i * 8;
        outT[(size_t)(n0 + nl) * K + k0 + tx] = tile[nl][tx];
    }
}

// s[i] = gamma*rsqrt(var+eps); o[i] = beta - mean*s
__global__ void k_bnprep(const float* __restrict__ gamma, const float* __restrict__ beta,
                         const float* __restrict__ mean, const float* __restrict__ var,
                         float* __restrict__ s, float* __restrict__ o, int n) {
    int i = blockIdx.x * blockDim.x + threadIdx.x;
    if (i >= n) return;
    float inv = gamma[i] * rsqrtf(var[i] + BN_EPS);
    s[i] = inv;
    o[i] = beta[i] - mean[i] * inv;
}

// ---------------- WMMA GEMM ----------------
// C = epilogue( A[M x K](bf16, row stride lda) @ Bt^T + bias ),
// Bt = B^T stored [Np x K](bf16, row stride K): both tiles stage as K-contiguous rows.
// Block tile 256(M) x 128(N), BK=32. 256 threads = 8 wave32 waves, 4(M) x 2(N),
// each wave computes 64x64 = 4x4 v_wmma_f32_16x16x32_bf16 tiles.
// Double-buffered LDS filled by global_load_async_to_lds_b128 (ASYNCcnt), so the
// next tile's HBM fetch overlaps the current tile's WMMAs with no register staging.
// EPI==0: +bias -> relu -> BN -> bf16 out. EPI==1: +bias -> BN -> tanh -> f32 out.

#define LDSTR 48   // LDS row stride in halves (96B: 16B aligned, 2-way-max conflicts)

union FragU { uint4 q[2]; v16bf v; };

template <int EPI, typename OutT>
__global__ __launch_bounds__(256)
void gemm_bf16(const u16t* __restrict__ A, int lda,
               const u16t* __restrict__ Bt,
               const float* __restrict__ bias, const float* __restrict__ s,
               const float* __restrict__ o, OutT* __restrict__ C, int ldc,
               int N, int K) {
    __shared__ u16t As[2][256 * LDSTR];
    __shared__ u16t Bs[2][128 * LDSTR];

    const int tid  = threadIdx.x;
    const int lane = tid & 31;          // wave32
    const int w    = tid >> 5;          // 0..7
    const int wm   = w >> 1;            // 0..3  -> M offset wm*64
    const int wn   = w & 1;             // 0..1  -> N offset wn*64
    const int rowBase = blockIdx.y * 256;
    const int colBase = blockIdx.x * 128;   // always < Np (padded), full tile valid

    const int lm   = lane & 15;
    const int lh   = lane >> 4;
    const int koff = lh ? 8 : 0;        // lanes 16-31 hold the other K-halves

    // staging-chunk coordinates (8 halves = 16B per chunk)
    const int ar[4] = { (tid + 0) >> 2, (tid + 256) >> 2, (tid + 512) >> 2, (tid + 768) >> 2 };
    const int akc   = (tid & 3) << 3;
    const int br[2] = { (tid + 0) >> 2, (tid + 256) >> 2 };

    v8f acc[4][4];
    v8f vz;
#pragma unroll
    for (int r = 0; r < 8; ++r) vz[r] = 0.0f;
#pragma unroll
    for (int mt = 0; mt < 4; ++mt)
#pragma unroll
        for (int nt = 0; nt < 4; ++nt) acc[mt][nt] = vz;

    const int KT = K >> 5;              // K multiple of 32

    // async-DMA one BK tile into LDS buffer nb (no VGPR staging)
    auto asyncTile = [&](int kt, int nb) {
        const int k0 = kt << 5;
#pragma unroll
        for (int i = 0; i < 4; ++i) {
            const u16t* gp = A + (size_t)(rowBase + ar[i]) * lda + k0 + akc;
            unsigned la = (unsigned)(size_t)&As[nb][ar[i] * LDSTR + akc];
            async_copy_b128(gp, la);
        }
#pragma unroll
        for (int i = 0; i < 2; ++i) {
            const u16t* gp = Bt + (size_t)(colBase + br[i]) * K + k0 + akc;
            unsigned la = (unsigned)(size_t)&Bs[nb][br[i] * LDSTR + akc];
            async_copy_b128(gp, la);
        }
    };

    // prologue: DMA tile 0 into buffer 0 and wait
    asyncTile(0, 0);
    async_wait0();

    for (int kt = 0; kt < KT; ++kt) {
        __syncthreads();                 // buf(kt&1) filled & visible; other buf free
        const int buf = kt & 1;

        // kick off next tile's DMA immediately; it overlaps the WMMAs below
        if (kt + 1 < KT) asyncTile(kt + 1, buf ^ 1);

        // fragments + WMMAs for current tile
        v16bf bfr[4];
#pragma unroll
        for (int nt = 0; nt < 4; ++nt) {
            const u16t* p = &Bs[buf][(wn * 64 + nt * 16 + lm) * LDSTR + koff];
            FragU u;
            u.q[0] = *(const uint4*)p;
            u.q[1] = *(const uint4*)(p + 16);
            bfr[nt] = u.v;
        }
#pragma unroll
        for (int mt = 0; mt < 4; ++mt) {
            const u16t* p = &As[buf][(wm * 64 + mt * 16 + lm) * LDSTR + koff];
            FragU u;
            u.q[0] = *(const uint4*)p;
            u.q[1] = *(const uint4*)(p + 16);
            v16bf afr = u.v;
#pragma unroll
            for (int nt = 0; nt < 4; ++nt)
                acc[mt][nt] = __builtin_amdgcn_wmma_f32_16x16x32_bf16(
                    false, afr, false, bfr[nt], (short)0, acc[mt][nt], false, false);
        }

        // my DMA into the other buffer must complete before the next barrier
        if (kt + 1 < KT) async_wait0();
    }

    // ---- fused epilogue ----
#pragma unroll
    for (int mt = 0; mt < 4; ++mt) {
#pragma unroll
        for (int nt = 0; nt < 4; ++nt) {
            int n = colBase + wn * 64 + nt * 16 + lm;
            if (n < N) {
                float bs = bias[n], sc = s[n], of = o[n];
#pragma unroll
                for (int r = 0; r < 8; ++r) {
                    int m = rowBase + wm * 64 + mt * 16 + lh * 8 + r;
                    float v = acc[mt][nt][r] + bs;
                    if constexpr (EPI == 0) {
                        v = fmaxf(v, 0.0f);          // relu
                        v = v * sc + of;             // BN
                        C[(size_t)m * ldc + n] = (OutT)f2bf(v);
                    } else {
                        v = v * sc + of;             // BN
                        C[(size_t)m * ldc + n] = (OutT)tanhf(v);
                    }
                }
            }
        }
    }
}

// ---------------- host launcher ----------------

extern "C" void kernel_launch(void* const* d_in, const int* in_sizes, int n_in,
                              void* d_out, int out_size, void* d_ws, size_t ws_size,
                              hipStream_t stream) {
    (void)in_sizes; (void)n_in; (void)out_size; (void)ws_size;
    const int B = 2048, K1 = 20000, N1 = 4000, N2 = 500;
    const int N1p = 4096, N2p = 512;     // padded to block-tile multiples

    const float* x   = (const float*)d_in[0];
    const float* m1  = (const float*)d_in[1];
    const float* m2  = (const float*)d_in[2];
    const float* W1  = (const float*)d_in[3];
    const float* b1  = (const float*)d_in[4];
    const float* W2  = (const float*)d_in[5];
    const float* b2  = (const float*)d_in[6];
    const float* g1  = (const float*)d_in[7];
    const float* be1 = (const float*)d_in[8];
    const float* mu1 = (const float*)d_in[9];
    const float* va1 = (const float*)d_in[10];
    const float* g2  = (const float*)d_in[11];
    const float* be2 = (const float*)d_in[12];
    const float* mu2 = (const float*)d_in[13];
    const float* va2 = (const float*)d_in[14];

    // workspace carve-up (256B aligned)
    char*  ws  = (char*)d_ws;
    size_t off = 0;
    auto carve = [&](size_t bytes) {
        size_t p = off;
        off = (off + bytes + 255) & ~(size_t)255;
        return p;
    };
    u16t*  xb   = (u16t*)(ws + carve((size_t)B * K1 * 2));     // bf16 x        [B x K1]
    u16t*  w1bT = (u16t*)(ws + carve((size_t)N1p * K1 * 2));   // bf16 (W1*m1)^T [N1p x K1]
    u16t*  w2bT = (u16t*)(ws + carve((size_t)N2p * N1 * 2));   // bf16 (W2*m2)^T [N2p x N1]
    u16t*  hb   = (u16t*)(ws + carve((size_t)B * N1p * 2));    // bf16 hidden    [B x N1p]
    float* s1   = (float*)(ws + carve((size_t)N1 * 4));
    float* o1   = (float*)(ws + carve((size_t)N1 * 4));
    float* s2   = (float*)(ws + carve((size_t)N2 * 4));
    float* o2   = (float*)(ws + carve((size_t)N2 * 4));

    // prep: fused mask-multiply + bf16 + transpose (zero-padded rows), x->bf16, BN affine
    {
        dim3 grid(N1p / 32, K1 / 32);
        k_mulcvt_T<<<grid, dim3(32, 8), 0, stream>>>(W1, m1, w1bT, K1, N1);
    }
    {
        dim3 grid(N2p / 32, N1 / 32);
        k_mulcvt_T<<<grid, dim3(32, 8), 0, stream>>>(W2, m2, w2bT, N1, N2);
    }
    {
        long long n4 = (long long)B * K1 / 4;
        k_cvt<<<(unsigned)((n4 + 255) / 256), 256, 0, stream>>>(x, xb, n4);
    }
    k_bnprep<<<(N1 + 255) / 256, 256, 0, stream>>>(g1, be1, mu1, va1, s1, o1, N1);
    k_bnprep<<<(N2 + 255) / 256, 256, 0, stream>>>(g2, be2, mu2, va2, s2, o2, N2);

    // layer 1: x @ W1m + b1 -> relu -> BN -> bf16 hidden (ldc = N1p)
    {
        dim3 grid(N1p / 128, B / 256);
        gemm_bf16<0, u16t><<<grid, 256, 0, stream>>>(xb, K1, w1bT, b1, s1, o1,
                                                     hb, N1p, N1, K1);
    }
    // layer 2: h @ W2m + b2 -> BN -> tanh -> f32 out (lda = N1p, ldc = N2)
    {
        dim3 grid(N2p / 128, B / 256);
        gemm_bf16<1, float><<<grid, 256, 0, stream>>>(hb, N1p, w2bT, b2, s2, o2,
                                                      (float*)d_out, N2, N2, N1);
    }
}
